// PTv3DownSample_26654567039611
// MI455X (gfx1250) — matrix operand order repaired and checked by reading, not compiled, not run on hardware
//
#include <hip/hip_runtime.h>
#include <hip/hip_bf16.h>
#include <float.h>

typedef __attribute__((ext_vector_type(16))) _Float16 v16h;
typedef __attribute__((ext_vector_type(8)))  _Float16 v8h;
typedef __attribute__((ext_vector_type(8)))  float    v8f;

#define B_CLOUDS 4
#define N_PER    8192
#define M_PER    4096
#define C_IN     64
#define C_OUT    128
#define N_TOT    (B_CLOUDS * N_PER)   // 32768
#define M_TOT    (B_CLOUDS * M_PER)   // 16384

// ---------------- prep kernels ----------------

__global__ void k_conv_x(const float* __restrict__ x, _Float16* __restrict__ xh, int n4) {
    int i = blockIdx.x * blockDim.x + threadIdx.x;
    if (i < n4) {
        float4 v = ((const float4*)x)[i];
        union { _Float16 h[4]; unsigned long long u; } u;
        u.h[0] = (_Float16)v.x; u.h[1] = (_Float16)v.y;
        u.h[2] = (_Float16)v.z; u.h[3] = (_Float16)v.w;
        ((unsigned long long*)xh)[i] = u.u;
    }
}

// Wp[n*64 + k] = (f16) W[k*128 + n]   (B matrix pre-transposed, K contiguous per column)
__global__ void k_pack_w(const float* __restrict__ W, _Float16* __restrict__ Wp) {
    int i = blockIdx.x * 256 + threadIdx.x;   // 8192
    int n = i >> 6, k = i & 63;
    Wp[i] = (_Float16)W[k * C_OUT + n];
}

__global__ void k_init(int* __restrict__ out_batch, float* __restrict__ stats) {
    int i = blockIdx.x * 256 + threadIdx.x;
    if (i < M_TOT) out_batch[i] = i >> 12;    // M_PER = 4096 = 2^12
    if (i < 256)   stats[i] = 0.0f;           // gsum[128] + gsumsq[128]
}

// ---------------- FPS: one block per cloud, pos + dists in LDS ----------------

__global__ __launch_bounds__(1024, 1)
void k_fps(const float* __restrict__ pos, float* __restrict__ out_pos) {
    extern __shared__ char sm[];
    float* dist = (float*)sm;
    float* px = dist + N_PER;
    float* py = px + N_PER;
    float* pz = py + N_PER;
    unsigned long long* wred = (unsigned long long*)(pz + N_PER);
    int* bcast = (int*)(wred + 32);

    const int b = blockIdx.x;
    const int tid = threadIdx.x;
    const int lane = tid & 31, wv = tid >> 5;
    const float* pb = pos + (size_t)b * N_PER * 3;

    for (int j = tid; j < N_PER; j += 1024) {
        px[j] = pb[j * 3 + 0];
        py[j] = pb[j * 3 + 1];
        pz[j] = pb[j * 3 + 2];
        dist[j] = FLT_MAX;
    }
    __syncthreads();

    float* op = out_pos + (size_t)b * M_PER * 3;
    if (tid == 0) { op[0] = px[0]; op[1] = py[0]; op[2] = pz[0]; }  // samp[0] = 0

    int last = 0;
    for (int i = 1; i < M_PER; ++i) {
        const float lx = px[last], ly = py[last], lz = pz[last];
        unsigned long long best = 0ull;
        #pragma unroll
        for (int t = 0; t < N_PER / 1024; ++t) {
            int j = t * 1024 + tid;
            float dx = px[j] - lx, dy = py[j] - ly, dz = pz[j] - lz;
            float d = dx * dx + dy * dy + dz * dz;
            float nd = fminf(dist[j], d);
            dist[j] = nd;
            // pack (dist, ~idx): max picks largest dist, ties -> smallest idx (jnp.argmax)
            unsigned long long key =
                ((unsigned long long)__float_as_uint(nd) << 32) | (unsigned)(~j);
            best = key > best ? key : best;
        }
        #pragma unroll
        for (int m = 16; m; m >>= 1) {
            unsigned long long o = __shfl_xor(best, m, 32);
            best = o > best ? o : best;
        }
        if (lane == 0) wred[wv] = best;
        __syncthreads();
        if (wv == 0) {
            unsigned long long k = wred[lane];
            #pragma unroll
            for (int m = 16; m; m >>= 1) {
                unsigned long long o = __shfl_xor(k, m, 32);
                k = o > k ? o : k;
            }
            if (lane == 0) {
                int j = (int)(~(unsigned)k);
                *bcast = j;
                op[i * 3 + 0] = px[j];
                op[i * 3 + 1] = py[j];
                op[i * 3 + 2] = pz[j];
            }
        }
        __syncthreads();
        last = *bcast;
    }
}

// ---------------- GEMM h = x@W + b via v_wmma_f32_16x16x32_f16, fused BN partials ----------------

__global__ __launch_bounds__(256)
void k_gemm(const _Float16* __restrict__ xh, const _Float16* __restrict__ Wp,
            const float* __restrict__ bias, float* __restrict__ h,
            float* __restrict__ gsum, float* __restrict__ gsumsq) {
    const int lane = threadIdx.x & 31;
    const int wv = threadIdx.x >> 5;      // 0..7 -> N tile
    const int n0 = wv * 16;
    const int c = lane & 15;
    const int g = lane >> 4;
    const float bv = bias[n0 + c];

    // B operand layout: lanes 0-15 -> K 0-15, lanes 16-31 -> K 16-31, col = lane&15
    v16h b1 = *(const v16h*)(Wp + (size_t)(n0 + c) * 64 + g * 16);
    v16h b2 = *(const v16h*)(Wp + (size_t)(n0 + c) * 64 + 32 + g * 16);

    float s1 = 0.0f, s2 = 0.0f;
    for (int t = 0; t < 16; ++t) {
        const int m0 = (blockIdx.x + t * 128) * 16;
        const _Float16* ap = xh + (size_t)(m0 + c) * 64 + g * 16;
        // A layout (16-bit 16x32): element s = K g*16+s (contiguous, matches B)
        v8h lo1 = *(const v8h*)(ap);
        v8h hi1 = *(const v8h*)(ap + 8);
        v8h lo2 = *(const v8h*)(ap + 32);
        v8h hi2 = *(const v8h*)(ap + 40);
        v16h a1, a2;
        #pragma unroll
        for (int i = 0; i < 8; ++i) {
            a1[i] = lo1[i]; a1[i + 8] = hi1[i];
            a2[i] = lo2[i]; a2[i + 8] = hi2[i];
        }
        v8f acc = {};
        acc = __builtin_amdgcn_wmma_f32_16x16x32_f16(false, a1, false, b1,
                                                     (short)0, acc, false, false);
        acc = __builtin_amdgcn_wmma_f32_16x16x32_f16(false, a2, false, b2,
                                                     (short)0, acc, false, false);
        // D layout: vgpr i -> (M = m0 + i + 8g, N = n0 + c)
        float* hp = h + (size_t)(m0 + 8 * g) * C_OUT + n0 + c;
        #pragma unroll
        for (int i = 0; i < 8; ++i) {
            float v = acc[i] + bv;
            hp[(size_t)i * C_OUT] = v;
            s1 += v;
            s2 += v * v;
        }
    }
    s1 += __shfl_xor(s1, 16, 32);
    s2 += __shfl_xor(s2, 16, 32);
    if (g == 0) {
        atomicAdd(&gsum[n0 + c], s1);
        atomicAdd(&gsumsq[n0 + c], s2);
    }
}

// ---------------- BN stats finalize + BN/GELU apply ----------------

__global__ void k_bnstats(const float* __restrict__ gsum, const float* __restrict__ gsumsq,
                          const float* __restrict__ gamma, const float* __restrict__ beta,
                          float* __restrict__ sc, float* __restrict__ sh) {
    int c = threadIdx.x;
    float mean = gsum[c] * (1.0f / N_TOT);
    float var = gsumsq[c] * (1.0f / N_TOT) - mean * mean;
    float s = gamma[c] * rsqrtf(var + 1e-5f);
    sc[c] = s;
    sh[c] = beta[c] - mean * s;
}

__device__ __forceinline__ float gelu_exact(float z) {
    return 0.5f * z * (1.0f + erff(z * 0.70710678118654752f));
}

__global__ __launch_bounds__(256)
void k_bngelu(float* __restrict__ h, const float* __restrict__ sc,
              const float* __restrict__ sh) {
    int i = blockIdx.x * 256 + threadIdx.x;   // float4 index, 1,048,576 total
    float4 v = ((float4*)h)[i];
    float4 s = ((const float4*)sc)[i & 31];
    float4 t = ((const float4*)sh)[i & 31];
    float4 o;
    o.x = gelu_exact(v.x * s.x + t.x);
    o.y = gelu_exact(v.y * s.y + t.y);
    o.z = gelu_exact(v.z * s.z + t.z);
    o.w = gelu_exact(v.w * s.w + t.w);
    ((float4*)h)[i] = o;
}

// ---------------- fused kNN top-16 + gather + channel max: one wave per query ----------------

__global__ __launch_bounds__(256)
void k_knnmax(const float* __restrict__ pos, const float* __restrict__ qpos,
              const float* __restrict__ h, float* __restrict__ out_x) {
    const int lane = threadIdx.x & 31;
    const int q = blockIdx.x * 8 + (threadIdx.x >> 5);
    const int b = q >> 12;
    const float* pb = pos + (size_t)b * N_PER * 3;
    const float qx = qpos[q * 3 + 0], qy = qpos[q * 3 + 1], qz = qpos[q * 3 + 2];

    float d[16];
    int id[16];
    #pragma unroll
    for (int s = 0; s < 16; ++s) { d[s] = FLT_MAX; id[s] = 0; }

    for (int t = 0; t < N_PER / 32; ++t) {
        int j = t * 32 + lane;
        float dx = pb[j * 3 + 0] - qx;
        float dy = pb[j * 3 + 1] - qy;
        float dz = pb[j * 3 + 2] - qz;
        float v = dx * dx + dy * dy + dz * dz;
        if (v < d[15]) {   // sorted-insert, fully unrolled (registers only)
            #pragma unroll
            for (int s = 15; s >= 1; --s) {
                bool up = v < d[s - 1];
                bool here = (!up) && (v < d[s]);
                float nd = up ? d[s - 1] : (here ? v : d[s]);
                int ni = up ? id[s - 1] : (here ? j : id[s]);
                d[s] = nd; id[s] = ni;
            }
            if (v < d[0]) { d[0] = v; id[0] = j; }
        }
    }

    // merge 32 sorted lists via wave-min on heads; fuse feature gather + max
    const float* hb = h + (size_t)b * N_PER * C_OUT + lane * 4;
    float4 acc = make_float4(-FLT_MAX, -FLT_MAX, -FLT_MAX, -FLT_MAX);
    #pragma unroll
    for (int r = 0; r < 16; ++r) {
        unsigned long long key =
            ((unsigned long long)__float_as_uint(d[0]) << 32) | (unsigned)lane;
        #pragma unroll
        for (int m = 16; m; m >>= 1) {
            unsigned long long o = __shfl_xor(key, m, 32);
            key = o < key ? o : key;
        }
        int wl = (int)(unsigned)key;          // winning lane
        int widx = __shfl(id[0], wl, 32);     // broadcast winning point index
        float4 hv = *(const float4*)(hb + (size_t)widx * C_OUT);
        acc.x = fmaxf(acc.x, hv.x);
        acc.y = fmaxf(acc.y, hv.y);
        acc.z = fmaxf(acc.z, hv.z);
        acc.w = fmaxf(acc.w, hv.w);
        if (lane == wl) {                     // pop head, shift (constant indices)
            #pragma unroll
            for (int s = 0; s < 15; ++s) { d[s] = d[s + 1]; id[s] = id[s + 1]; }
            d[15] = FLT_MAX;
        }
    }
    *(float4*)(out_x + (size_t)q * C_OUT + lane * 4) = acc;
}

// ---------------- launch ----------------

extern "C" void kernel_launch(void* const* d_in, const int* in_sizes, int n_in,
                              void* d_out, int out_size, void* d_ws, size_t ws_size,
                              hipStream_t stream) {
    const float* x     = (const float*)d_in[0];
    const float* pos   = (const float*)d_in[1];
    // d_in[2] = batch (int32), not needed: segments are contiguous by construction
    const float* W     = (const float*)d_in[3];
    const float* bias  = (const float*)d_in[4];
    const float* gamma = (const float*)d_in[5];
    const float* beta  = (const float*)d_in[6];

    float* out_x   = (float*)d_out;                      // [16384, 128]
    float* out_pos = out_x + (size_t)M_TOT * C_OUT;      // [16384, 3]
    int*   out_bat = (int*)(out_pos + (size_t)M_TOT * 3);// [16384]

    char* ws = (char*)d_ws;
    _Float16* xh    = (_Float16*)ws;                             // 4 MB
    _Float16* Wp    = (_Float16*)(ws + 4 * 1024 * 1024);         // 16 KB
    float*    stats = (float*)(ws + 4 * 1024 * 1024 + 16 * 1024);// 2 KB
    float* gsum = stats, *gsumsq = stats + 128, *sc = stats + 256, *sh = stats + 384;
    float*    h     = (float*)(ws + 4 * 1024 * 1024 + 32 * 1024);// 16 MB

    k_conv_x<<<2048, 256, 0, stream>>>(x, xh, N_TOT * C_IN / 4);
    k_pack_w<<<32, 256, 0, stream>>>(W, Wp);
    k_init<<<64, 256, 0, stream>>>(out_bat, stats);

    size_t fps_lds = (size_t)N_PER * 4 * 4 + 32 * 8 + 16;        // dists + xyz + reduce + bcast
    k_fps<<<4, 1024, fps_lds, stream>>>(pos, out_pos);

    k_gemm<<<128, 256, 0, stream>>>(xh, Wp, bias, h, gsum, gsumsq);
    k_bnstats<<<1, 128, 0, stream>>>(gsum, gsumsq, gamma, beta, sc, sh);
    k_bngelu<<<4096, 256, 0, stream>>>(h, sc, sh);

    k_knnmax<<<2048, 256, 0, stream>>>(pos, out_pos, h, out_x);
}
